// CrossAttentionSkip_63333587747556
// MI455X (gfx1250) — matrix operand order repaired and checked by reading, compiled, not run on hardware
//
#include <hip/hip_runtime.h>
#include <hip/hip_bf16.h>
#include <math.h>

// ---------------------------------------------------------------------------
// CDNA5 (gfx1250) cross-attention block, bf16 WMMA everywhere.
// Shapes: B=2, S=4096, C_enc=512, C_dec=256, heads=8, hd=32, d_ff=1024.
// ---------------------------------------------------------------------------

typedef __attribute__((ext_vector_type(16))) __bf16 v16bf;
typedef __attribute__((ext_vector_type(8)))  __bf16 bf16x8;
typedef __attribute__((ext_vector_type(8)))  float  v8f;

#define DEV __device__ __forceinline__

static constexpr int S_LEN = 4096;
static constexpr int BATCH = 2;
static constexpr int CENC  = 512;
static constexpr int CDEC  = 256;
static constexpr int NHEAD = 8;
static constexpr int HDIM  = 32;
static constexpr int DFF   = 1024;
static constexpr int ROWS  = BATCH * S_LEN;   // 8192

DEV unsigned short f2bf_bits(float x) {
  union { float f; unsigned u; } a; a.f = x;
  unsigned r = (a.u + 0x7FFFu + ((a.u >> 16) & 1u)) >> 16;
  return (unsigned short)r;
}
DEV __bf16 f2bf(float x) {
  union { unsigned short s; __bf16 b; } o; o.s = f2bf_bits(x); return o.b;
}
DEV float bf2f(__bf16 x) {
  union { unsigned short s; __bf16 b; } i; i.b = x;
  union { unsigned u; float f; } o; o.u = ((unsigned)i.s) << 16; return o.f;
}

// Lane-exchange via ds_swizzle (group-of-32, xor_mask in offset[14:10],
// and_mask=0x1f): one LDS-HW op, no index VGPR, no canonicalization.
template <int MASK>
DEV float swz_xor_f(float x) {
  int r = __builtin_amdgcn_ds_swizzle((int)__float_as_uint(x),
                                      (MASK << 10) | 0x1f);
  return __uint_as_float((unsigned)r);
}
// Reduce across the 16 lanes of each half-wave (rows of a WMMA f32 C tile
// live entirely inside one 16-lane half; masks 1/2/4/8 stay inside it).
DEV float half_reduce_max(float t) {
  t = fmaxf(t, swz_xor_f<1>(t));
  t = fmaxf(t, swz_xor_f<2>(t));
  t = fmaxf(t, swz_xor_f<4>(t));
  t = fmaxf(t, swz_xor_f<8>(t));
  return t;
}
DEV float half_reduce_sum(float t) {
  t += swz_xor_f<1>(t);
  t += swz_xor_f<2>(t);
  t += swz_xor_f<4>(t);
  t += swz_xor_f<8>(t);
  return t;
}

// A-fragment (16x32 bf16, M x K), row-major source with leading dim lda.
// Layout (ISA 7.12.2): lanes 0-15 row M=lane, elems 0..7 -> K 0..7, 8..15 -> K 16..23;
// lanes 16-31 row M=lane-16, elems 0..7 -> K 8..15, 8..15 -> K 24..31.
DEV v16bf load_a_frag(const __bf16* base, int lda, int lane) {
  int r = lane & 15, h = lane >> 4;
  const __bf16* p = base + (size_t)r * lda + h * 8;
  union { v16bf v; bf16x8 q[2]; } u;
  u.q[0] = *(const bf16x8*)(p);
  u.q[1] = *(const bf16x8*)(p + 16);
  return u.v;
}

// B-fragment (32x16 bf16, K x N) where the source is laid out N-major:
// element (K=k, N=c) at base[c*ldb + k].  Lane c = column, lanes 0-15 K=0..7,16..23,
// lanes 16-31 K=8..15,24..31 (same K order as the A-fragment).
DEV v16bf load_b_frag(const __bf16* base, int ldb, int lane) {
  int c = lane & 15, h = lane >> 4;
  const __bf16* p = base + (size_t)c * ldb + h * 8;
  union { v16bf v; bf16x8 q[2]; } u;
  u.q[0] = *(const bf16x8*)(p);
  u.q[1] = *(const bf16x8*)(p + 16);
  return u.v;
}

DEV v8f wmma_bf16(v16bf a, v16bf b, v8f c) {
  return __builtin_amdgcn_wmma_f32_16x16x32_bf16(false, a, false, b, (short)0, c,
                                                 false, false);
}

// ---------------------------------------------------------------------------
// LayerNorm over channel dim for (B, C, S) tensors -> bf16 (B*S, C) rows.
// ---------------------------------------------------------------------------
__global__ void ln_chan_kernel(const float* __restrict__ src,
                               const float* __restrict__ g,
                               const float* __restrict__ bb,
                               __bf16* __restrict__ dst, int C) {
  int row = blockIdx.x * blockDim.x + threadIdx.x;
  if (row >= ROWS) return;
  int b = row >> 12, s = row & (S_LEN - 1);
  const float* base = src + (size_t)b * C * S_LEN + s;
  float sum = 0.f, sq = 0.f;
  for (int c = 0; c < C; ++c) {
    float v = base[(size_t)c * S_LEN];
    sum += v; sq += v * v;
  }
  float inv = 1.0f / (float)C;
  float mean = sum * inv;
  float var = sq * inv - mean * mean;
  float rstd = rsqrtf(var + 1e-5f);
  __bf16* drow = dst + (size_t)row * C;
  for (int c = 0; c < C; ++c) {
    float v = base[(size_t)c * S_LEN];
    drow[c] = f2bf((v - mean) * rstd * g[c] + bb[c]);
  }
}

// LayerNorm over contiguous rows of 256 f32 -> bf16 rows (one wave per row).
__global__ void ln_row_kernel(const float* __restrict__ x,
                              const float* __restrict__ g,
                              const float* __restrict__ bb,
                              __bf16* __restrict__ out) {
  int wid = (int)((blockIdx.x * blockDim.x + threadIdx.x) >> 5);
  int lane = threadIdx.x & 31;
  if (wid >= ROWS) return;
  const float* row = x + (size_t)wid * CDEC;
  float s = 0.f, s2 = 0.f;
#pragma unroll
  for (int i = 0; i < 8; ++i) {
    float v = row[lane + 32 * i];
    s += v; s2 += v * v;
  }
  // full-wave reduction: masks 1..16
  s += swz_xor_f<1>(s);  s2 += swz_xor_f<1>(s2);
  s += swz_xor_f<2>(s);  s2 += swz_xor_f<2>(s2);
  s += swz_xor_f<4>(s);  s2 += swz_xor_f<4>(s2);
  s += swz_xor_f<8>(s);  s2 += swz_xor_f<8>(s2);
  s += swz_xor_f<16>(s); s2 += swz_xor_f<16>(s2);
  float mean = s * (1.f / 256.f);
  float var = s2 * (1.f / 256.f) - mean * mean;
  float rstd = rsqrtf(var + 1e-5f);
  __bf16* orow = out + (size_t)wid * CDEC;
#pragma unroll
  for (int i = 0; i < 8; ++i) {
    int c = lane + 32 * i;
    orow[c] = f2bf((row[c] - mean) * rstd * g[c] + bb[c]);
  }
}

// Transpose + convert: src f32 (K x N) -> dst bf16 (N x K).
__global__ void tc_kernel(const float* __restrict__ src, __bf16* __restrict__ dst,
                          int K, int N) {
  int idx = blockIdx.x * blockDim.x + threadIdx.x;
  if (idx >= N * K) return;
  int n = idx / K, k = idx - n * K;
  dst[idx] = f2bf(src[(size_t)k * N + n]);
}

// V (B*S, 256) bf16 -> Vt (B, H, hd, S) bf16 so P@V B-fragments are contiguous.
__global__ void vt_kernel(const __bf16* __restrict__ V, __bf16* __restrict__ Vt) {
  int idx = blockIdx.x * blockDim.x + threadIdx.x;
  if (idx >= BATCH * NHEAD * HDIM * S_LEN) return;
  int s = idx & (S_LEN - 1);
  int t = idx >> 12;
  int d = t & 31; t >>= 5;
  int h = t & 7;  t >>= 3;
  int b = t;
  Vt[idx] = V[((size_t)(b * S_LEN) + s) * CDEC + h * HDIM + d];
}

// ---------------------------------------------------------------------------
// Generic WMMA GEMM: C(MxN) = A(MxK bf16) @ Wt(NxK bf16)^T + bias, epilogues:
//  MODE 0: bf16 out.   MODE 1: bf16 out with exact GELU.
//  MODE 2: f32 out = acc + bias + decoder_feat[b, c, s]   (residual 1)
//  MODE 3: f32 out transposed to (B,C,S) = acc + bias + x[row, c] (residual 2)
// One wave computes a 32x64 tile (2x4 WMMA tiles); block = 4 waves = 128x64.
// ---------------------------------------------------------------------------
template <int MODE>
DEV void gemm_epilogue(const v8f& c, int mrow0, int ncol0, int lane,
                       const float* __restrict__ bias, __bf16* __restrict__ outB,
                       float* __restrict__ outF, const float* __restrict__ resid,
                       int N) {
  int col = ncol0 + (lane & 15);
  int half = lane >> 4;
  float bv = bias[col];
#pragma unroll
  for (int r = 0; r < 8; ++r) {
    int row = mrow0 + r + 8 * half;
    float v = c[r] + bv;
    if (MODE == 0) {
      outB[(size_t)row * N + col] = f2bf(v);
    } else if (MODE == 1) {
      float gl = 0.5f * v * (1.0f + erff(v * 0.70710678118f));
      outB[(size_t)row * N + col] = f2bf(gl);
    } else if (MODE == 2) {
      int b = row >> 12, s = row & (S_LEN - 1);
      float res = resid[((size_t)b * CDEC + col) * S_LEN + s];
      outF[(size_t)row * CDEC + col] = v + res;
    } else {  // MODE 3
      int b = row >> 12, s = row & (S_LEN - 1);
      float res = resid[(size_t)row * CDEC + col];
      outF[((size_t)b * CDEC + col) * S_LEN + s] = v + res;
    }
  }
}

template <int MODE>
__global__ void __launch_bounds__(128)
gemm_kernel(const __bf16* __restrict__ A, const __bf16* __restrict__ Wt,
            const float* __restrict__ bias, __bf16* __restrict__ outB,
            float* __restrict__ outF, const float* __restrict__ resid,
            int M, int N, int K) {
  int lane = threadIdx.x & 31;
  int wave = threadIdx.x >> 5;
  int n0 = blockIdx.x * 64;
  int m0 = (blockIdx.y * 4 + wave) * 32;
  if (m0 >= M || n0 >= N) return;

  v8f acc[8];
  const v8f z = {};
#pragma unroll
  for (int i = 0; i < 8; ++i) acc[i] = z;

  // K is a multiple of 64 for every call site; unroll 2 K-steps so two
  // independent load/WMMA groups interleave and hide the WMMA hazard slots.
#pragma unroll 2
  for (int kk = 0; kk < K; kk += 32) {
    // cover the next K-step's panel while this one computes
    __builtin_prefetch(A + (size_t)(m0 + (lane & 15)) * K + kk + 32, 0, 3);
    __builtin_prefetch(A + (size_t)(m0 + 16 + (lane & 15)) * K + kk + 32, 0, 3);
    __builtin_prefetch(Wt + (size_t)(n0 + lane) * K + kk + 32, 0, 3);

    v16bf a0 = load_a_frag(A + (size_t)m0 * K + kk, K, lane);
    v16bf a1 = load_a_frag(A + (size_t)(m0 + 16) * K + kk, K, lane);
    v16bf bfr[4];
#pragma unroll
    for (int j = 0; j < 4; ++j)
      bfr[j] = load_b_frag(Wt + (size_t)(n0 + 16 * j) * K + kk, K, lane);
#pragma unroll
    for (int j = 0; j < 4; ++j) {
      acc[j]     = wmma_bf16(a0, bfr[j], acc[j]);
      acc[4 + j] = wmma_bf16(a1, bfr[j], acc[4 + j]);
    }
  }
#pragma unroll
  for (int j = 0; j < 4; ++j) {
    gemm_epilogue<MODE>(acc[j],     m0,      n0 + 16 * j, lane, bias, outB, outF, resid, N);
    gemm_epilogue<MODE>(acc[4 + j], m0 + 16, n0 + 16 * j, lane, bias, outB, outF, resid, N);
  }
}

// ---------------------------------------------------------------------------
// Flash attention: one wave per (b, h, 16-query tile).  hd=32 == WMMA K, so one
// WMMA per 16x16 score tile.  64-key chunks: online-softmax reductions are
// amortized over 4 score tiles.  exp() is computed IN PLACE on the score tiles
// to keep live ranges (and register copies) down.  P goes through per-wave LDS
// to be re-laid out as A-fragments for the P@V WMMAs.
// ---------------------------------------------------------------------------
__global__ void __launch_bounds__(128)
attn_kernel(const __bf16* __restrict__ Q, const __bf16* __restrict__ Km,
            const __bf16* __restrict__ Vt, __bf16* __restrict__ O) {
  __shared__ __bf16 plds[4][16 * 64];
  int lane = threadIdx.x & 31;
  int wave = threadIdx.x >> 5;
  int wid = blockIdx.x * 4 + wave;               // 0 .. 4095
  int qt = wid & 255;
  int h = (wid >> 8) & 7;
  int b = wid >> 11;
  int q0 = qt * 16;

  // Q fragment, pre-scaled by hd^-0.5 = 0.176776695
  const __bf16* Qbase = Q + ((size_t)(b * S_LEN) + q0) * CDEC + h * HDIM;
  union { v16bf v; bf16x8 q[2]; __bf16 e[16]; } qa;
  {
    int r = lane & 15, hf = lane >> 4;
    const __bf16* p = Qbase + (size_t)r * CDEC + hf * 8;
    qa.q[0] = *(const bf16x8*)(p);
    qa.q[1] = *(const bf16x8*)(p + 16);
#pragma unroll
    for (int i = 0; i < 16; ++i) qa.e[i] = f2bf(bf2f(qa.e[i]) * 0.17677669529f);
  }

  float mr[8], lr[8];
#pragma unroll
  for (int i = 0; i < 8; ++i) { mr[i] = -INFINITY; lr[i] = 0.f; }
  v8f o0 = {}, o1 = {};
  const v8f zc = {};
  int half = lane >> 4;
  int col = lane & 15;
  __bf16* pl = plds[wave];
  const __bf16* Vhead = Vt + ((size_t)(b * NHEAD + h) * HDIM) * S_LEN;

  for (int c0 = 0; c0 < S_LEN; c0 += 64) {
    const __bf16* Kbase = Km + ((size_t)(b * S_LEN) + c0) * CDEC + h * HDIM;
    v8f s0, s1, s2, s3;
    {
      v16bf k0 = load_b_frag(Kbase, CDEC, lane);
      v16bf k1 = load_b_frag(Kbase + 16 * CDEC, CDEC, lane);
      v16bf k2 = load_b_frag(Kbase + 32 * CDEC, CDEC, lane);
      v16bf k3 = load_b_frag(Kbase + 48 * CDEC, CDEC, lane);
      s0 = wmma_bf16(qa.v, k0, zc);
      s1 = wmma_bf16(qa.v, k1, zc);
      s2 = wmma_bf16(qa.v, k2, zc);
      s3 = wmma_bf16(qa.v, k3, zc);
    }

    // Per-row online softmax; exp in place on the score tiles.
#pragma unroll
    for (int i = 0; i < 8; ++i) {
      float t = fmaxf(fmaxf(s0[i], s1[i]), fmaxf(s2[i], s3[i]));
      t = half_reduce_max(t);
      float mn = fmaxf(mr[i], t);
      float alpha = __expf(mr[i] - mn);
      mr[i] = mn;
      s0[i] = __expf(s0[i] - mn);
      s1[i] = __expf(s1[i] - mn);
      s2[i] = __expf(s2[i] - mn);
      s3[i] = __expf(s3[i] - mn);
      float rs = half_reduce_sum((s0[i] + s1[i]) + (s2[i] + s3[i]));
      lr[i] = lr[i] * alpha + rs;
      o0[i] *= alpha;
      o1[i] *= alpha;
    }
    // P (16x64) -> LDS row-major, then reload as two A-fragments.
#pragma unroll
    for (int i = 0; i < 8; ++i) {
      int row = i + 8 * half;
      pl[row * 64 + col]      = f2bf(s0[i]);
      pl[row * 64 + 16 + col] = f2bf(s1[i]);
      pl[row * 64 + 32 + col] = f2bf(s2[i]);
      pl[row * 64 + 48 + col] = f2bf(s3[i]);
    }
    v16bf pa0 = load_a_frag(pl, 64, lane);        // keys c0 .. c0+31
    v16bf pa1 = load_a_frag(pl + 32, 64, lane);   // keys c0+32 .. c0+63
    const __bf16* Vbase = Vhead + c0;
    v16bf v0 = load_b_frag(Vbase, S_LEN, lane);              // dims 0..15, k 0..31
    v16bf v1 = load_b_frag(Vbase + 16 * S_LEN, S_LEN, lane); // dims 16..31, k 0..31
    v16bf v2 = load_b_frag(Vbase + 32, S_LEN, lane);         // dims 0..15, k 32..63
    v16bf v3 = load_b_frag(Vbase + 16 * S_LEN + 32, S_LEN, lane);
    o0 = wmma_bf16(pa0, v0, o0);
    o1 = wmma_bf16(pa0, v1, o1);
    o0 = wmma_bf16(pa1, v2, o0);
    o1 = wmma_bf16(pa1, v3, o1);
  }

  __bf16* Obase = O + ((size_t)(b * S_LEN) + q0) * CDEC + h * HDIM;
#pragma unroll
  for (int i = 0; i < 8; ++i) {
    float inv = 1.0f / lr[i];
    int row = i + 8 * half;
    Obase[(size_t)row * CDEC + col] = f2bf(o0[i] * inv);
    Obase[(size_t)row * CDEC + 16 + col] = f2bf(o1[i] * inv);
  }
}

// ---------------------------------------------------------------------------
extern "C" void kernel_launch(void* const* d_in, const int* in_sizes, int n_in,
                              void* d_out, int out_size, void* d_ws, size_t ws_size,
                              hipStream_t stream) {
  (void)in_sizes; (void)n_in; (void)out_size; (void)ws_size;
  const float* enc   = (const float*)d_in[0];
  const float* dec   = (const float*)d_in[1];
  const float* Wq    = (const float*)d_in[2];
  const float* bq    = (const float*)d_in[3];
  const float* Wk    = (const float*)d_in[4];
  const float* bk    = (const float*)d_in[5];
  const float* Wv    = (const float*)d_in[6];
  const float* bv    = (const float*)d_in[7];
  const float* Wo    = (const float*)d_in[8];
  const float* bo    = (const float*)d_in[9];
  const float* g_enc = (const float*)d_in[10];
  const float* b_enc = (const float*)d_in[11];
  const float* g_dec = (const float*)d_in[12];
  const float* b_dec = (const float*)d_in[13];
  const float* g_out = (const float*)d_in[14];
  const float* b_out = (const float*)d_in[15];
  const float* W1    = (const float*)d_in[16];
  const float* b1    = (const float*)d_in[17];
  const float* W2    = (const float*)d_in[18];
  const float* b2    = (const float*)d_in[19];
  float* out = (float*)d_out;

  char* ws = (char*)d_ws;
  size_t off = 0;
  auto alloc = [&](size_t bytes) { char* p = ws + off; off += (bytes + 255) & ~size_t(255); return p; };
  __bf16* enc_n = (__bf16*)alloc((size_t)ROWS * CENC * 2);
  __bf16* dec_n = (__bf16*)alloc((size_t)ROWS * CDEC * 2);
  __bf16* Qb    = (__bf16*)alloc((size_t)ROWS * CDEC * 2);
  __bf16* Kb    = (__bf16*)alloc((size_t)ROWS * CDEC * 2);
  __bf16* Vb    = (__bf16*)alloc((size_t)ROWS * CDEC * 2);
  __bf16* Vtb   = (__bf16*)alloc((size_t)ROWS * CDEC * 2);
  __bf16* aOut  = (__bf16*)alloc((size_t)ROWS * CDEC * 2);
  float*  x     = (float*) alloc((size_t)ROWS * CDEC * 4);
  __bf16* xln   = (__bf16*)alloc((size_t)ROWS * CDEC * 2);
  __bf16* ffn   = (__bf16*)alloc((size_t)ROWS * DFF  * 2);
  __bf16* WqT   = (__bf16*)alloc((size_t)CDEC * CDEC * 2);
  __bf16* WkT   = (__bf16*)alloc((size_t)CDEC * CENC * 2);
  __bf16* WvT   = (__bf16*)alloc((size_t)CDEC * CENC * 2);
  __bf16* WoT   = (__bf16*)alloc((size_t)CDEC * CDEC * 2);
  __bf16* W1T   = (__bf16*)alloc((size_t)DFF  * CDEC * 2);
  __bf16* W2T   = (__bf16*)alloc((size_t)CDEC * DFF  * 2);

  // 1) LayerNorms over channel dim (bf16 sequences).
  ln_chan_kernel<<<ROWS / 256, 256, 0, stream>>>(enc, g_enc, b_enc, enc_n, CENC);
  ln_chan_kernel<<<ROWS / 256, 256, 0, stream>>>(dec, g_dec, b_dec, dec_n, CDEC);

  // 2) Weight transpose+convert (K x N f32 -> N x K bf16).
  auto tc = [&](const float* src, __bf16* dst, int K, int N) {
    int n = N * K;
    tc_kernel<<<(n + 255) / 256, 256, 0, stream>>>(src, dst, K, N);
  };
  tc(Wq, WqT, CDEC, CDEC);
  tc(Wk, WkT, CENC, CDEC);
  tc(Wv, WvT, CENC, CDEC);
  tc(Wo, WoT, CDEC, CDEC);
  tc(W1, W1T, CDEC, DFF);
  tc(W2, W2T, DFF, CDEC);

  // 3) Q/K/V projections (WMMA).
  dim3 blk(128);
  gemm_kernel<0><<<dim3(CDEC / 64, ROWS / 128), blk, 0, stream>>>(
      dec_n, WqT, bq, Qb, nullptr, nullptr, ROWS, CDEC, CDEC);
  gemm_kernel<0><<<dim3(CDEC / 64, ROWS / 128), blk, 0, stream>>>(
      enc_n, WkT, bk, Kb, nullptr, nullptr, ROWS, CDEC, CENC);
  gemm_kernel<0><<<dim3(CDEC / 64, ROWS / 128), blk, 0, stream>>>(
      enc_n, WvT, bv, Vb, nullptr, nullptr, ROWS, CDEC, CENC);

  // 4) V -> per-head transposed layout for contiguous P@V B-fragments.
  {
    int n = BATCH * NHEAD * HDIM * S_LEN;
    vt_kernel<<<(n + 255) / 256, 256, 0, stream>>>(Vb, Vtb);
  }

  // 5) Flash attention: 2*8*256 = 4096 waves, 4 waves/block.
  attn_kernel<<<1024, 128, 0, stream>>>(Qb, Kb, Vtb, aOut);

  // 6) Output projection + residual (f32 x).
  gemm_kernel<2><<<dim3(CDEC / 64, ROWS / 128), blk, 0, stream>>>(
      aOut, WoT, bo, nullptr, x, dec, ROWS, CDEC, CDEC);

  // 7) LayerNorm(x) -> bf16.
  ln_row_kernel<<<ROWS / 8, 256, 0, stream>>>(x, g_out, b_out, xln);

  // 8) FFN1 with exact GELU.
  gemm_kernel<1><<<dim3(DFF / 64, ROWS / 128), blk, 0, stream>>>(
      xln, W1T, b1, ffn, nullptr, nullptr, ROWS, DFF, CDEC);

  // 9) FFN2 + residual, written transposed into (B, C, S) output.
  gemm_kernel<3><<<dim3(CDEC / 64, ROWS / 128), blk, 0, stream>>>(
      ffn, W2T, b2, nullptr, out, x, ROWS, CDEC, DFF);
}
